// CCBlock_81020263072504
// MI455X (gfx1250) — matrix operand order repaired and checked by reading, not compiled, not run on hardware
//
#include <hip/hip_runtime.h>

// ---------------- problem constants ----------------
#define Bn  8
#define Cn  256
#define Hn  128
#define Wn  128
#define CRn 32
#define HWn (Hn*Wn)

// ---------------- workspace layout (bf16 elements) ----------------
#define QOFF  0
#define KOFF  (Bn*HWn*CRn)              /* 4,194,304  */
#define VOFF  (2*KOFF)
#define EOFF  (3*KOFF)                  /* energy_H / attnH probs */
#define OWOFF (EOFF + Bn*Hn*Wn*Hn)      /* width-agg partial      */
#define OHOFF (OWOFF + Bn*HWn*CRn)      /* height-agg partial     */
// total bf16 elements = 37,748,736 -> ~75.5 MB workspace

typedef __attribute__((ext_vector_type(16))) __bf16 v16bf;
typedef __attribute__((ext_vector_type(8)))  __bf16 v8bf;
typedef __attribute__((ext_vector_type(8)))  float  v8f;

union FragU { v16bf v; v8bf h[2]; };

__device__ __forceinline__ v8f zero8() {
  v8f z; for (int i = 0; i < 8; ++i) z[i] = 0.0f; return z;
}

__device__ __forceinline__ uint32_t pack_bf16(float lo, float hi) {
  union { __bf16 b; uint16_t u; } a, c;
  a.b = (__bf16)lo; c.b = (__bf16)hi;
  return (uint32_t)a.u | ((uint32_t)c.u << 16);
}

// low 32 bits of a generic pointer to a __shared__ object = wave-relative LDS
// byte address (FLAT LDS aperture)
#define LDSOFF(p) ((uint32_t)(uintptr_t)(p))

// ---- CDNA5 async global<->LDS copies (ASYNCcnt) ----
__device__ __forceinline__ void async_load_b128(uint32_t lds_off, const void* gptr) {
  asm volatile("global_load_async_to_lds_b128 %0, %1, off"
               :: "v"(lds_off), "v"((uint64_t)(uintptr_t)gptr) : "memory");
}
__device__ __forceinline__ void async_store_b128(void* gptr, uint32_t lds_off) {
  asm volatile("global_store_async_from_lds_b128 %0, %1, off"
               :: "v"((uint64_t)(uintptr_t)gptr), "v"(lds_off) : "memory");
}
__device__ __forceinline__ void wait_async() {
  asm volatile("s_wait_asynccnt 0x0" ::: "memory");
}

// A fragment (16x32 bf16, M x K), source row-major with K contiguous.
__device__ __forceinline__ v16bf frag_A(const __bf16* base, int stride, int lane) {
  const int m  = lane & 15;
  const int kb = (lane >> 4) << 3;
  const __bf16* p = base + m * stride + kb;
  FragU f;
  f.h[0] = *(const v8bf*)(p);
  f.h[1] = *(const v8bf*)(p + 16);
  return f.v;
}

// B fragment (32x16 bf16, K x N), source stored column-major: [n][k] K-contiguous.
__device__ __forceinline__ v16bf frag_B(const __bf16* baseT, int stride, int lane) {
  const int nn = lane & 15;
  const int kb = (lane >> 4) << 3;
  const __bf16* p = baseT + nn * stride + kb;
  FragU f;
  f.h[0] = *(const v8bf*)(p);
  f.h[1] = *(const v8bf*)(p + 16);
  return f.v;
}

#define WMMA_BF16(a, b, c) \
  __builtin_amdgcn_wmma_f32_16x16x32_bf16(false, (a), false, (b), (short)0, (c), false, false)

// ============================================================================
// K1: q/k/v projection. GEMM M=96 (q|k|v stacked), K=256, N=pixels.
// Output layout per buffer: (b, h, w, c) with c contiguous (bf16).
// ============================================================================
__global__ void __launch_bounds__(256) k_qkv(
    const float* __restrict__ x,
    const float* __restrict__ Wq, const float* __restrict__ bq,
    const float* __restrict__ Wk, const float* __restrict__ bk,
    const float* __restrict__ Wv, const float* __restrict__ bv,
    __bf16* __restrict__ ws) {
  __shared__ alignas(16) __bf16 ldsW[96 * 256];   // 48 KB  (row-major, K contig)
  __shared__ alignas(16) __bf16 ldsX[128 * 40];   // 10 KB  (pixel-major, padded)
  const int t    = threadIdx.x;
  const int lane = t & 31;
  const int wvid = t >> 5;
  const int pg0  = blockIdx.x * 128;     // block of 128 pixels, never straddles b
  const int b    = pg0 / HWn;
  const int p0   = pg0 % HWn;

  // stage stacked weights Wq|Wk|Wv as bf16
  for (int idx = t; idx < 96 * 256; idx += 256) {
    const int r = idx >> 8, c = idx & 255;
    float w = (r < 32) ? Wq[r * 256 + c]
            : (r < 64) ? Wk[(r - 32) * 256 + c]
                       : Wv[(r - 64) * 256 + c];
    ldsW[idx] = (__bf16)w;
  }

  v8f acc[6];
  for (int i = 0; i < 6; ++i) acc[i] = zero8();
  __syncthreads();

  const int ch2   = (t & 15) * 2;   // channel pair handled by this thread
  const int chunk = t >> 4;         // 8-pixel chunk (0..15)
  uint32_t* lx32  = (uint32_t*)ldsX;

  for (int kk = 0; kk < 8; ++kk) {            // K loop: 8 x 32 channels
    const int c0 = kk * 32;
    {   // stage x tile transposed & packed: ldsX[pixel][c], 32 ch x 128 px
      const float* s0 = x + ((size_t)b * Cn + (c0 + ch2)) * HWn + p0 + chunk * 8;
      const float* s1 = s0 + HWn;
      if (kk < 7) __builtin_prefetch(s0 + (size_t)32 * HWn, 0, 1);
      for (int j = 0; j < 8; j += 4) {
        const float4 a4 = *(const float4*)(s0 + j);
        const float4 b4 = *(const float4*)(s1 + j);
        lx32[(chunk * 8 + j + 0) * 20 + (t & 15)] = pack_bf16(a4.x, b4.x);
        lx32[(chunk * 8 + j + 1) * 20 + (t & 15)] = pack_bf16(a4.y, b4.y);
        lx32[(chunk * 8 + j + 2) * 20 + (t & 15)] = pack_bf16(a4.z, b4.z);
        lx32[(chunk * 8 + j + 3) * 20 + (t & 15)] = pack_bf16(a4.w, b4.w);
      }
    }
    __syncthreads();
    const v16bf bf = frag_B(ldsX + (wvid * 16) * 40, 40, lane);
    for (int mt = 0; mt < 6; ++mt) {
      const v16bf af = frag_A(ldsW + (mt * 16) * 256 + c0, 256, lane);
      acc[mt] = WMMA_BF16(af, bf, acc[mt]);
    }
    __syncthreads();
  }

  // store: rows 0-31 -> q, 32-63 -> k, 64-95 -> v ; packed b32 stores
  const int n   = lane & 15;
  const int mb  = (lane >> 4) * 8;
  const int pix = p0 + wvid * 16 + n;
  for (int mt = 0; mt < 6; ++mt) {
    const int sel = mt >> 1;                       // 0=q 1=k 2=v
    __bf16* dst = ws + (size_t)sel * KOFF;
    const float* bias = (sel == 0) ? bq : (sel == 1) ? bk : bv;
    const size_t base = ((size_t)b * HWn + pix) * CRn;
    for (int r = 0; r < 8; r += 2) {
      const int c = ((mt & 1) ? 16 : 0) + mb + r;
      *(uint32_t*)(dst + base + c) =
          pack_bf16(acc[mt][r] + bias[c], acc[mt][r + 1] + bias[c + 1]);
    }
  }
}

// ============================================================================
// K2: energy_H per (b,w). Eh[h][i] = sum_c q[b,h,w,c]*k[b,i,w,c], K=32.
// -inf on the h==i diagonal. Stored bf16, layout (b, h, w, i).
// ============================================================================
__global__ void __launch_bounds__(256) k_energy_h(__bf16* __restrict__ ws) {
  __shared__ alignas(16) __bf16 ldsQ[128 * 32];   // 8 KB
  __shared__ alignas(16) __bf16 ldsK[128 * 32];   // 8 KB
  const int t = threadIdx.x, lane = t & 31, wvid = t >> 5;
  const int b = blockIdx.x / Wn, w = blockIdx.x % Wn;
  const __bf16* q = ws + QOFF;
  const __bf16* k = ws + KOFF;
  __bf16* e = ws + EOFF;
  {   // async staged copy: one b128 per thread per buffer
    const int hh = t >> 1, hv = t & 1;
    const size_t off = ((size_t)b * HWn + hh * Wn + w) * CRn + hv * 16;
    const uint32_t l = (hh * 32 + hv * 16) * 2;  // byte offset in tile
    async_load_b128(LDSOFF(ldsQ) + l, q + off);
    async_load_b128(LDSOFF(ldsK) + l, k + off);
  }
  wait_async();
  __syncthreads();
  const int mt = wvid;                         // h tile owned by this wave
  const v16bf af = frag_A(ldsQ + (mt * 16) * 32, 32, lane);
  const int n = lane & 15, mb = (lane >> 4) * 8;
  for (int nt = 0; nt < 8; ++nt) {
    const v16bf bf = frag_B(ldsK + (nt * 16) * 32, 32, lane);
    v8f acc = zero8();
    acc = WMMA_BF16(af, bf, acc);
    for (int r = 0; r < 8; ++r) {
      const int hh = mt * 16 + mb + r;
      const int ii = nt * 16 + n;
      const float vvv = (hh == ii) ? -__builtin_inff() : acc[r];
      e[(((size_t)b * Hn + hh) * Wn + w) * Hn + ii] = (__bf16)vvv;
    }
  }
}

// ============================================================================
// K3: per (b,h) row: energy_W GEMM -> LDS logits next to energy_H block,
// fused softmax over H+W=256, write attnH probs back to ws (in place over
// energy_H), width aggregation GEMM outW[c,w] = sum_j v[c,j]*aW[w,j].
// ============================================================================
#define LR 264   // logit row stride (bf16 elems), 528 B = 16B-aligned
__global__ void __launch_bounds__(256) k_row_softmax_wagg(__bf16* __restrict__ ws) {
  __shared__ alignas(16) __bf16 ldsQ[128 * 32];    // 8 KB
  __shared__ alignas(16) __bf16 ldsK[128 * 32];    // 8 KB
  __shared__ alignas(16) __bf16 ldsVT[32 * 128];   // 8 KB  v row transposed [c][j]
  __shared__ alignas(16) __bf16 logit[128 * LR];   // 66 KB [w][0..127]=Eh [128..255]=Ew
  const int t = threadIdx.x, lane = t & 31, wvid = t >> 5;
  const int b = blockIdx.x / Hn, h = blockIdx.x % Hn;
  const __bf16* q = ws + QOFF;
  const __bf16* k = ws + KOFF;
  const __bf16* v = ws + VOFF;
  __bf16* e  = ws + EOFF;
  __bf16* ow = ws + OWOFF;

  const int px = t >> 1, hv = t & 1;
  {   // async: q/k rows into LDS + energy_H block into logits[w][0..127]
    const size_t off = ((size_t)b * HWn + h * Wn + px) * CRn + hv * 16;
    const uint32_t l = (px * 32 + hv * 16) * 2;
    async_load_b128(LDSOFF(ldsQ) + l, q + off);
    async_load_b128(LDSOFF(ldsK) + l, k + off);
    const __bf16* esrc = e + ((size_t)b * Hn + h) * Wn * Hn;
    for (int idx = t * 8; idx < 128 * 128; idx += 2048) {
      const int wI = idx >> 7, iI = idx & 127;
      async_load_b128(LDSOFF(&logit[wI * LR + iI]), esrc + idx);
    }
  }
  {   // meanwhile: transpose v row manually (scatter, cannot async)
    const size_t off = ((size_t)b * HWn + h * Wn + px) * CRn + hv * 16;
    const uint4 vv = *(const uint4*)(v + off);
    const __bf16* vb = (const __bf16*)&vv;
    for (int j = 0; j < 16; ++j) ldsVT[(hv * 16 + j) * 128 + px] = vb[j];
  }
  wait_async();
  __syncthreads();
  {   // energy_W GEMM: M=N=128, K=32 ; wave owns one 16-row tile of w
    const v16bf af = frag_A(ldsQ + (wvid * 16) * 32, 32, lane);
    const int n = lane & 15, mb = (lane >> 4) * 8;
    for (int nt = 0; nt < 8; ++nt) {
      const v16bf bf = frag_B(ldsK + (nt * 16) * 32, 32, lane);
      v8f acc = zero8();
      acc = WMMA_BF16(af, bf, acc);
      for (int r = 0; r < 8; ++r)
        logit[(wvid * 16 + mb + r) * LR + Hn + nt * 16 + n] = (__bf16)acc[r];
    }
  }
  __syncthreads();
  {   // softmax over 256 logits per w: thread pair (t, t^1) splits the row
    const int wI = t >> 1;
    __bf16* row = &logit[wI * LR + hv * 128];
    float m = -__builtin_inff();
    for (int j = 0; j < 128; ++j) m = fmaxf(m, (float)row[j]);
    m = fmaxf(m, __shfl_xor(m, 1));
    float s = 0.0f;
    for (int j = 0; j < 128; ++j) s += __expf((float)row[j] - m);
    s += __shfl_xor(s, 1);
    const float inv = 1.0f / s;
    for (int j = 0; j < 128; ++j)
      row[j] = (__bf16)(__expf((float)row[j] - m) * inv);
  }
  __syncthreads();
  {   // async write attnH probs back over the energy_H buffer (same layout)
    __bf16* edst = e + ((size_t)b * Hn + h) * Wn * Hn;
    for (int idx = t * 8; idx < 128 * 128; idx += 2048) {
      const int wI = idx >> 7, iI = idx & 127;
      async_store_b128(edst + idx, LDSOFF(&logit[wI * LR + iI]));
    }
  }
  {   // width aggregation: M=32(c) N=128(w) K=128(j); wave owns one w tile
    const int nt = wvid;
    v8f a0 = zero8(), a1 = zero8();
    for (int ks = 0; ks < 4; ++ks) {
      const v16bf bf = frag_B(logit + (nt * 16) * LR + Hn + ks * 32, LR, lane);
      const v16bf f0 = frag_A(ldsVT + ks * 32, 128, lane);
      const v16bf f1 = frag_A(ldsVT + 16 * 128 + ks * 32, 128, lane);
      a0 = WMMA_BF16(f0, bf, a0);
      a1 = WMMA_BF16(f1, bf, a1);
    }
    const int n = lane & 15, mb = (lane >> 4) * 8;
    const int wI = nt * 16 + n;
    const size_t base = ((size_t)b * HWn + h * Wn + wI) * CRn;
    for (int r = 0; r < 8; r += 2) {
      *(uint32_t*)(ow + base + mb + r)      = pack_bf16(a0[r], a0[r + 1]);
      *(uint32_t*)(ow + base + 16 + mb + r) = pack_bf16(a1[r], a1[r + 1]);
    }
  }
  wait_async();
}

// ============================================================================
// K4: height aggregation per (b,w): outH[c,h] = sum_i v[b,i,w,c]*aH[h,i]
// M=32(c) N=128(h) K=128(i); B fragments packed straight from global.
// ============================================================================
__global__ void __launch_bounds__(256) k_height_agg(__bf16* __restrict__ ws) {
  __shared__ alignas(16) __bf16 ldsVT[32 * 128];   // [c][i]
  const int t = threadIdx.x, lane = t & 31, wvid = t >> 5;
  const int b = blockIdx.x / Wn, w = blockIdx.x % Wn;
  const __bf16* v = ws + VOFF;
  const __bf16* e = ws + EOFF;                     // attnH probs now
  __bf16* oh = ws + OHOFF;
  {
    const int px = t >> 1, hv = t & 1;
    const uint4 vv = *(const uint4*)(v + ((size_t)b * HWn + px * Wn + w) * CRn + hv * 16);
    const __bf16* vb = (const __bf16*)&vv;
    for (int j = 0; j < 16; ++j) ldsVT[(hv * 16 + j) * 128 + px] = vb[j];
  }
  __syncthreads();
  const int nt = wvid;                             // h tile
  const __bf16* ebase = e + (((size_t)b * Hn) * Wn + w) * Hn;  // + h*W*H + i
  v8f a0 = zero8(), a1 = zero8();
  for (int ks = 0; ks < 4; ++ks) {
    const v16bf bf = frag_B(ebase + (size_t)(nt * 16) * Wn * Hn + ks * 32, Wn * Hn, lane);
    const v16bf f0 = frag_A(ldsVT + ks * 32, 128, lane);
    const v16bf f1 = frag_A(ldsVT + 16 * 128 + ks * 32, 128, lane);
    a0 = WMMA_BF16(f0, bf, a0);
    a1 = WMMA_BF16(f1, bf, a1);
  }
  const int n = lane & 15, mb = (lane >> 4) * 8;
  const int hI = nt * 16 + n;
  const size_t base = ((size_t)b * HWn + hI * Wn + w) * CRn;
  for (int r = 0; r < 8; r += 2) {
    *(uint32_t*)(oh + base + mb + r)      = pack_bf16(a0[r], a0[r + 1]);
    *(uint32_t*)(oh + base + 16 + mb + r) = pack_bf16(a1[r], a1[r + 1]);
  }
}

// ============================================================================
// K5: z projection + bias + residual. M=256, K=32, N=pixels.
// ============================================================================
__global__ void __launch_bounds__(256) k_zproj(
    const float* __restrict__ x, const float* __restrict__ Wz,
    const float* __restrict__ bz, const __bf16* __restrict__ ws,
    float* __restrict__ out) {
  __shared__ alignas(16) __bf16 ldsWz[256 * 32];   // 16 KB (row-major K contig)
  __shared__ alignas(16) __bf16 ldsS[128 * 32];    // 8 KB  (pixel-major)
  const int t = threadIdx.x, lane = t & 31, wvid = t >> 5;
  const int pg0 = blockIdx.x * 128;
  const int b = pg0 / HWn, p0 = pg0 % HWn;
  const __bf16* ow = ws + OWOFF;
  const __bf16* oh = ws + OHOFF;
  for (int idx = t; idx < 256 * 32; idx += 256) ldsWz[idx] = (__bf16)Wz[idx];
  {   // sum the two aggregation partials into the B tile
    const int px = t >> 1, hv = t & 1;
    const size_t off = ((size_t)b * HWn + p0 + px) * CRn + hv * 16;
    const uint4 ua = *(const uint4*)(ow + off);
    const uint4 ub = *(const uint4*)(oh + off);
    const __bf16* pa = (const __bf16*)&ua;
    const __bf16* pb = (const __bf16*)&ub;
    alignas(16) __bf16 sum[16];
    for (int j = 0; j < 16; ++j) sum[j] = (__bf16)((float)pa[j] + (float)pb[j]);
    *(uint4*)&ldsS[px * 32 + hv * 16] = *(const uint4*)sum;
  }
  __syncthreads();
  const v16bf bf = frag_B(ldsS + (wvid * 16) * 32, 32, lane);
  const int n = lane & 15, mb = (lane >> 4) * 8;
  const int pix = p0 + wvid * 16 + n;
  for (int mt = 0; mt < 16; ++mt) {
    const v16bf af = frag_A(ldsWz + (mt * 16) * 32, 32, lane);
    v8f acc = zero8();
    acc = WMMA_BF16(af, bf, acc);
    for (int r = 0; r < 8; ++r) {
      const int co = mt * 16 + mb + r;
      const size_t o = ((size_t)b * Cn + co) * HWn + pix;
      out[o] = acc[r] + bz[co] + x[o];
    }
  }
}

// ============================================================================
extern "C" void kernel_launch(void* const* d_in, const int* in_sizes, int n_in,
                              void* d_out, int out_size, void* d_ws, size_t ws_size,
                              hipStream_t stream) {
  (void)in_sizes; (void)n_in; (void)out_size; (void)ws_size;
  const float* x  = (const float*)d_in[0];
  const float* Wq = (const float*)d_in[1];
  const float* bq = (const float*)d_in[2];
  const float* Wk = (const float*)d_in[3];
  const float* bk = (const float*)d_in[4];
  const float* Wv = (const float*)d_in[5];
  const float* bv = (const float*)d_in[6];
  const float* Wz = (const float*)d_in[7];
  const float* bz = (const float*)d_in[8];
  __bf16* ws = (__bf16*)d_ws;
  float* out = (float*)d_out;
  const dim3 blk(256);
  k_qkv             <<<dim3(Bn * HWn / 128), blk, 0, stream>>>(x, Wq, bq, Wk, bk, Wv, bv, ws);
  k_energy_h        <<<dim3(Bn * Wn),        blk, 0, stream>>>(ws);
  k_row_softmax_wagg<<<dim3(Bn * Hn),        blk, 0, stream>>>(ws);
  k_height_agg      <<<dim3(Bn * Wn),        blk, 0, stream>>>(ws);
  k_zproj           <<<dim3(Bn * HWn / 128), blk, 0, stream>>>(x, Wz, bz, ws, out);
}
